// DPAM_1391569404259
// MI455X (gfx1250) — matrix operand order repaired and checked by reading, not compiled, not run on hardware
//
#include <hip/hip_runtime.h>

// ---------------- types ----------------
typedef __attribute__((ext_vector_type(16))) __bf16        v16bf;
typedef __attribute__((ext_vector_type(8)))  float         v8f;
typedef __attribute__((ext_vector_type(4)))  unsigned int  u32x4;
typedef __attribute__((ext_vector_type(8)))  unsigned int  u32x8;
typedef __attribute__((ext_vector_type(4)))  float         f32x4;

#define LOG2E 1.44269504088896340736f

// truncate-to-bf16 (hot paths): 1 VALU op
__device__ __forceinline__ unsigned int f2bf(float f) {
  return __float_as_uint(f) >> 16;
}
// RNE convert (cold weight prep only)
__device__ __forceinline__ unsigned int f2bf_rne(float f) {
  unsigned int u = __float_as_uint(f);
  return (u + 0x7FFFu + ((u >> 16) & 1u)) >> 16;
}
// pack two bf16 with a single v_perm_b32: D = {hi16(b), hi16(a)}
__device__ __forceinline__ unsigned int pk2(float a, float b) {
  return __builtin_amdgcn_perm(__float_as_uint(b), __float_as_uint(a), 0x07060302u);
}

union FragU { u32x4 q[2]; u32x8 u8; v16bf v; };

__device__ __forceinline__ v8f zero8() {
  v8f z;
#pragma unroll
  for (int i = 0; i < 8; i++) z[i] = 0.0f;
  return z;
}

__device__ __forceinline__ v8f wmma_bf16(v16bf a, v16bf b, v8f c) {
  return __builtin_amdgcn_wmma_f32_16x16x32_bf16(false, a, false, b, (short)0, c,
                                                 false, false);
}

// A operand: 16x32 bf16; lane = row (lane&15); lanes 0-15 hold K {0..15},
// lanes 16-31 hold K {16..31}.  p -> 32-element row.
__device__ __forceinline__ v16bf load_frag_row32(const unsigned short* p, int h) {
  FragU f;
  f.q[0] = *(const u32x4*)(p + 16 * h);
  f.q[1] = *(const u32x4*)(p + 16 * h + 8);
  return f.v;
}

// B operand: 32x16 bf16; lane = column n (lane&15); half h holds K=16h..16h+15.
__device__ __forceinline__ v16bf load_frag_col32(const unsigned short* p, int h) {
  FragU f;
  f.q[0] = *(const u32x4*)(p + 16 * h);
  f.q[1] = *(const u32x4*)(p + 16 * h + 8);
  return f.v;
}

__device__ __forceinline__ u32x4 pack8(v8f a) {
  u32x4 r;
  r[0] = pk2(a[0], a[1]);
  r[1] = pk2(a[2], a[3]);
  r[2] = pk2(a[4], a[5]);
  r[3] = pk2(a[6], a[7]);
  return r;
}

// ---- gfx1250 async global->LDS DMA (ASYNCcnt path) ----
__device__ __forceinline__ void async_b128(unsigned short* ldsp, const unsigned short* gp) {
  unsigned int l = (unsigned int)(unsigned long long)ldsp;  // low 32b = LDS byte offset
  asm volatile("global_load_async_to_lds_b128 %0, %1, off" :: "v"(l), "v"(gp) : "memory");
}
__device__ __forceinline__ void wait_async0() {
  asm volatile("s_wait_asynccnt 0x0" ::: "memory");
}

// ---- gfx1250 Tensor Data Mover: 2D tile [tile1 x tile0] from row-major tensor ----
typedef __attribute__((ext_vector_type(4))) unsigned int su4;
typedef __attribute__((ext_vector_type(8))) unsigned int su8;

// load tile of 64 keys x 256 cols from d^T ([256][4096] bf16) into contiguous LDS
__device__ __forceinline__ void tdm_load_dtile(unsigned short* ldsdst,
                                               const unsigned short* gsrc) {
  unsigned long long ga = (unsigned long long)gsrc;
  unsigned int lds = (unsigned int)(unsigned long long)ldsdst;
  su4 g0;
  g0[0] = 1u;                                            // count=1 (valid user D#)
  g0[1] = lds;                                           // lds_addr (bytes)
  g0[2] = (unsigned int)ga;                              // global_addr[31:0]
  g0[3] = (unsigned int)((ga >> 32) & 0x01FFFFFFull)     // global_addr[56:32]
          | 0x80000000u;                                 // type=2 ("image")
  su8 g1;
  g1[0] = 0x00010000u;        // workgroup_mask=0, data_size=1 (2 bytes)
  g1[1] = 4096u << 16;        // tensor_dim0[15:0]=4096 (keys)
  g1[2] = 256u << 16;         // tensor_dim0[31:16]=0 ; tensor_dim1[15:0]=256 (cols)
  g1[3] = 64u << 16;          // tensor_dim1[31:16]=0 ; tile_dim0=64
  g1[4] = 256u;               // tile_dim1=256 ; tile_dim2=0
  g1[5] = 4096u;              // tensor_dim0_stride[31:0]=4096
  g1[6] = 0u;                 // stride hi / dim1_stride lo
  g1[7] = 0u;
  su4 gz;
  gz[0] = gz[1] = gz[2] = gz[3] = 0u;
  asm volatile("tensor_load_to_lds %0, %1, %2, %3"
               :: "s"(g0), "s"(g1), "s"(gz), "s"(gz) : "memory");
}
__device__ __forceinline__ void wait_tensor0() {
  asm volatile("s_wait_tensorcnt 0x0" ::: "memory");
}

// ---------------- constants ----------------
#define NPIX  4096
#define CCH   256

// workspace layout (ushort elements); weights are first, contiguous
#define OFF_WBT   0u
#define OFF_WCT   8192u
#define OFF_WB2T  16384u
#define OFF_WC2T  24576u
#define OFF_WDT   32768u                 // 256*256
#define W_TOTAL   98304u                 // total weight ushorts
#define OFF_B     98304u                 // [B,N,32]
#define OFF_C     (OFF_B  + 1048576u)
#define OFF_B2    (OFF_C  + 1048576u)
#define OFF_C2    (OFF_B2 + 1048576u)
#define OFF_DT    (OFF_C2 + 1048576u)    // [B,256,N]

// ---------------- kernel 1: weight convert + transpose ----------------
__global__ __launch_bounds__(256) void prep_weights(
    const float* __restrict__ Wb, const float* __restrict__ Wc,
    const float* __restrict__ Wb2, const float* __restrict__ Wc2,
    const float* __restrict__ Wd, unsigned short* __restrict__ ws) {
  int t = blockIdx.x * 256 + threadIdx.x;      // 0 .. 65535
  int o = t >> 8, k = t & 255;
  if (t < 32 * 256) {
    ws[OFF_WBT  + o * 256 + k] = (unsigned short)f2bf_rne(Wb [k * 32 + o]);
    ws[OFF_WCT  + o * 256 + k] = (unsigned short)f2bf_rne(Wc [k * 32 + o]);
    ws[OFF_WB2T + o * 256 + k] = (unsigned short)f2bf_rne(Wb2[k * 32 + o]);
    ws[OFF_WC2T + o * 256 + k] = (unsigned short)f2bf_rne(Wc2[k * 32 + o]);
  }
  ws[OFF_WDT + o * 256 + k] = (unsigned short)f2bf_rne(Wd[k * 256 + o]);
}

// Build bf16 A fragments for x1/x2 from interleaved x[...,2].
__device__ __forceinline__ void load_x_frags(const f32x4* xrow, int kc, int h,
                                             v16bf* a1, v16bf* a2) {
  FragU f1, f2;
  int base = kc * 16;
#pragma unroll
  for (int c = 0; c < 2; c++) {
    int idx = base + c * 4 + 8 * h;
#pragma unroll
    for (int j = 0; j < 4; j++) {
      f32x4 t = xrow[idx + j];
      f1.q[c][j] = pk2(t[0], t[2]);
      f2.q[c][j] = pk2(t[1], t[3]);
    }
  }
  *a1 = f1.v;
  *a2 = f2.v;
}

// ---------------- kernel 2: projections ----------------
// grid: 256 blocks (batch*32 tiles of 128 rows), 256 threads; weights staged in LDS.
__global__ __launch_bounds__(256) void proj_kernel(
    const float* __restrict__ x, unsigned short* __restrict__ ws) {
  extern __shared__ unsigned short smem[];   // 98304 ushorts = 192KB (all weights)
  unsigned short* bqw  = ws + OFF_B;
  unsigned short* cqw  = ws + OFF_C;
  unsigned short* b2qw = ws + OFF_B2;
  unsigned short* c2qw = ws + OFF_C2;
  unsigned short* dT   = ws + OFF_DT;

  int tid = threadIdx.x;
  int wave = tid >> 5, lane = tid & 31;
  int h = lane >> 4, m = lane & 15;
  int bi = blockIdx.x >> 5;
  int n0 = (blockIdx.x & 31) * 128 + wave * 16;
  long long rowg = (long long)bi * NPIX + n0 + m;
  const f32x4* xrow = (const f32x4*)x + rowg * 128;

  { // stage all weights into LDS (once per block)
    const u32x4* g = (const u32x4*)ws;
    u32x4* s = (u32x4*)smem;
    for (int i = tid; i < (int)(W_TOTAL / 8); i += 256) s[i] = g[i];
  }
  __syncthreads();
  const unsigned short* sWb  = smem + OFF_WBT;
  const unsigned short* sWc  = smem + OFF_WCT;
  const unsigned short* sWb2 = smem + OFF_WB2T;
  const unsigned short* sWc2 = smem + OFF_WC2T;
  const unsigned short* sWd  = smem + OFF_WDT;

  // ---- sweep 1: d = x2 @ Wd, stored transposed ----
  {
    v8f accd[16];
#pragma unroll
    for (int ct = 0; ct < 16; ct++) accd[ct] = zero8();
    for (int kc = 0; kc < 8; kc++) {
      v16bf a1, a2;
      load_x_frags(xrow, kc, h, &a1, &a2);
      // 2-deep B-fragment rotation: b0,b1 live together -> distinct registers,
      // so ds_loads stay >=2 WMMAs ahead of their consumer.
      v16bf b0 = load_frag_col32(sWd + m * 256 + kc * 32, h);
      v16bf b1 = load_frag_col32(sWd + (16 + m) * 256 + kc * 32, h);
#pragma unroll
      for (int ct = 0; ct < 16; ct += 2) {
        v16bf n0 = (ct < 14) ? load_frag_col32(sWd + ((ct + 2) * 16 + m) * 256 + kc * 32, h) : b0;
        accd[ct] = wmma_bf16(a2, b0, accd[ct]);
        v16bf n1 = (ct < 14) ? load_frag_col32(sWd + ((ct + 3) * 16 + m) * 256 + kc * 32, h) : b1;
        accd[ct + 1] = wmma_bf16(a2, b1, accd[ct + 1]);
        b0 = n0;
        b1 = n1;
      }
    }
#pragma unroll
    for (int ct = 0; ct < 16; ct++) {
      long long col = ct * 16 + m;
      *(u32x4*)(dT + ((long long)bi * 256 + col) * NPIX + n0 + 8 * h) = pack8(accd[ct]);
    }
  }

  // ---- sweep 2: b, c, b2, c2 ----
  {
    v8f aB[2], aC[2], aB2[2], aC2[2];
#pragma unroll
    for (int ct = 0; ct < 2; ct++) { aB[ct] = zero8(); aC[ct] = zero8(); aB2[ct] = zero8(); aC2[ct] = zero8(); }
    for (int kc = 0; kc < 8; kc++) {
      v16bf a1, a2;
      load_x_frags(xrow, kc, h, &a1, &a2);
#pragma unroll
      for (int ct = 0; ct < 2; ct++) {
        int co = (ct * 16 + m) * 256 + kc * 32;
        aB [ct] = wmma_bf16(a1, load_frag_col32(sWb  + co, h), aB [ct]);
        aC [ct] = wmma_bf16(a1, load_frag_col32(sWc  + co, h), aC [ct]);
        aB2[ct] = wmma_bf16(a2, load_frag_col32(sWb2 + co, h), aB2[ct]);
        aC2[ct] = wmma_bf16(a2, load_frag_col32(sWc2 + co, h), aC2[ct]);
      }
    }
#pragma unroll
    for (int ct = 0; ct < 2; ct++) {
#pragma unroll
      for (int v = 0; v < 8; v++) {
        long long pix = (long long)bi * NPIX + n0 + 8 * h + v;
        int ch = ct * 16 + m;
        bqw [pix * 32 + ch] = (unsigned short)f2bf(aB [ct][v]);
        cqw [pix * 32 + ch] = (unsigned short)f2bf(aC [ct][v]);
        b2qw[pix * 32 + ch] = (unsigned short)f2bf(aB2[ct][v]);
        c2qw[pix * 32 + ch] = (unsigned short)f2bf(aC2[ct][v]);
      }
    }
  }
}

// online softmax stat update over one 16x64 logit stripe
__device__ __forceinline__ void stat_update(const v8f S[4], float mm[8], float ll[8]) {
#pragma unroll
  for (int v = 0; v < 8; v++) {
    float tmax = fmaxf(fmaxf(S[0][v], S[1][v]), fmaxf(S[2][v], S[3][v]));
#pragma unroll
    for (int msk = 1; msk < 16; msk <<= 1) tmax = fmaxf(tmax, __shfl_xor(tmax, msk, 32));
    float mn = fmaxf(mm[v], tmax);
    float s = 0.0f;
#pragma unroll
    for (int ct = 0; ct < 4; ct++)
      s += __builtin_amdgcn_exp2f((S[ct][v] - mn) * LOG2E);
#pragma unroll
    for (int msk = 1; msk < 16; msk <<= 1) s += __shfl_xor(s, msk, 32);
    ll[v] = ll[v] * __builtin_amdgcn_exp2f((mm[v] - mn) * LOG2E) + s;
    mm[v] = mn;
  }
}

// issue async DMA of one 64x32 c-tile pair (per-thread slice)
__device__ __forceinline__ void issue_c_tile(unsigned short* b1, unsigned short* b2,
                                             const unsigned short* cqw,
                                             const unsigned short* c2qw,
                                             int bi, int kt, int cr, int cc) {
  long long src = ((long long)bi * NPIX + kt * 64 + cr) * 32 + cc;
  async_b128(b1 + cr * 32 + cc, cqw + src);
  async_b128(b2 + cr * 32 + cc, c2qw + src);
}

// ---------------- kernel 3: fused dual flash attention ----------------
// grid: 256 blocks (batch * 32 tiles of 128 query rows), 256 threads
__global__ __launch_bounds__(256) void attn_kernel(
    const float* __restrict__ x, const unsigned short* __restrict__ ws,
    const float* __restrict__ gamma, float* __restrict__ out) {
  extern __shared__ unsigned short smem[];   // 49152 ushorts = 96KB
  unsigned short* sC1 = smem;                // 2 x 2048
  unsigned short* sC2 = smem + 4096;         // 2 x 2048
  unsigned short* sD  = smem + 8192;         // 2 x 16384
  unsigned short* sP  = smem + 40960;        // 8192

  const unsigned short* bqw  = ws + OFF_B;
  const unsigned short* cqw  = ws + OFF_C;
  const unsigned short* b2qw = ws + OFF_B2;
  const unsigned short* c2qw = ws + OFF_C2;
  const unsigned short* dT   = ws + OFF_DT;

  int tid = threadIdx.x, wave = tid >> 5, lane = tid & 31;
  int h = lane >> 4, m = lane & 15;
  int bi = blockIdx.x >> 5;
  int qw = (blockIdx.x & 31) * 128 + wave * 16;
  long long qrow = (long long)bi * NPIX + qw + m;

  v16bf A1 = load_frag_row32(bqw  + qrow * 32, h);
  v16bf A2 = load_frag_row32(b2qw + qrow * 32, h);

  float m1[8], l1[8], m2[8], l2[8];
#pragma unroll
  for (int v = 0; v < 8; v++) { m1[v] = m2[v] = -1e30f; l1[v] = l2[v] = 0.0f; }

  int cr = tid >> 2, cc = (tid & 3) * 8;     // c-tile cooperative coords

  // ===== pass 1: softmax statistics (double-buffered c tiles) =====
  issue_c_tile(sC1, sC2, cqw, c2qw, bi, 0, cr, cc);
  wait_async0();
  __syncthreads();
  for (int kt = 0; kt < 64; kt++) {
    int cur = kt & 1;
    if (kt < 63)
      issue_c_tile(sC1 + (1 - cur) * 2048, sC2 + (1 - cur) * 2048, cqw, c2qw, bi, kt + 1, cr, cc);
    const unsigned short* c1 = sC1 + cur * 2048;
    const unsigned short* c2 = sC2 + cur * 2048;
    v8f S1[4], S2[4];
#pragma unroll
    for (int ct = 0; ct < 4; ct++) {
      S1[ct] = wmma_bf16(A1, load_frag_col32(c1 + (ct * 16 + m) * 32, h), zero8());
      S2[ct] = wmma_bf16(A2, load_frag_col32(c2 + (ct * 16 + m) * 32, h), zero8());
    }
    stat_update(S1, m1, l1);
    stat_update(S2, m2, l2);
    wait_async0();
    __syncthreads();
  }
  float k1[8], i1[8], k2[8], i2[8];
#pragma unroll
  for (int v = 0; v < 8; v++) {
    k1[v] = m1[v] * LOG2E; i1[v] = 1.0f / l1[v];
    k2[v] = m2[v] * LOG2E; i2[v] = 1.0f / l2[v];
  }

  // ===== pass 2: P = exp1/l1 + exp2/l2 ;  O += P @ d (double-buffered) =====
  v8f acc[16];
#pragma unroll
  for (int ct = 0; ct < 16; ct++) acc[ct] = zero8();
  unsigned short* pw = sP + wave * 16 * 64;

  issue_c_tile(sC1, sC2, cqw, c2qw, bi, 0, cr, cc);
  if (wave == 0) tdm_load_dtile(sD, dT + (long long)bi * 256 * NPIX);
  wait_async0();
  if (wave == 0) wait_tensor0();
  __syncthreads();
  for (int kt = 0; kt < 64; kt++) {
    int cur = kt & 1;
    if (kt < 63) {
      issue_c_tile(sC1 + (1 - cur) * 2048, sC2 + (1 - cur) * 2048, cqw, c2qw, bi, kt + 1, cr, cc);
      if (wave == 0)
        tdm_load_dtile(sD + (1 - cur) * 16384,
                       dT + (long long)bi * 256 * NPIX + (kt + 1) * 64);
    }
    const unsigned short* c1 = sC1 + cur * 2048;
    const unsigned short* c2 = sC2 + cur * 2048;
    const unsigned short* dc = sD + cur * 16384;

    // logits + normalized probabilities -> sP
#pragma unroll
    for (int ct = 0; ct < 4; ct++) {
      v8f S1 = wmma_bf16(A1, load_frag_col32(c1 + (ct * 16 + m) * 32, h), zero8());
      v8f S2 = wmma_bf16(A2, load_frag_col32(c2 + (ct * 16 + m) * 32, h), zero8());
#pragma unroll
      for (int v = 0; v < 8; v++) {
        float e = __builtin_amdgcn_exp2f(S1[v] * LOG2E - k1[v]) * i1[v] +
                  __builtin_amdgcn_exp2f(S2[v] * LOG2E - k2[v]) * i2[v];
        pw[(v + 8 * h) * 64 + ct * 16 + m] = (unsigned short)f2bf(e);
      }
    }

    // O += P @ d (2-deep pipelined B-frag rotation)
#pragma unroll
    for (int kh = 0; kh < 2; kh++) {
      v16bf AP = load_frag_row32(pw + m * 64 + kh * 32, h);
      v16bf b0 = load_frag_col32(dc + m * 64 + kh * 32, h);
      v16bf b1 = load_frag_col32(dc + (16 + m) * 64 + kh * 32, h);
#pragma unroll
      for (int ct = 0; ct < 16; ct += 2) {
        v16bf n0 = (ct < 14) ? load_frag_col32(dc + ((ct + 2) * 16 + m) * 64 + kh * 32, h) : b0;
        acc[ct] = wmma_bf16(AP, b0, acc[ct]);
        v16bf n1 = (ct < 14) ? load_frag_col32(dc + ((ct + 3) * 16 + m) * 64 + kh * 32, h) : b1;
        acc[ct + 1] = wmma_bf16(AP, b1, acc[ct + 1]);
        b0 = n0;
        b1 = n1;
      }
    }
    wait_async0();
    if (wave == 0) wait_tensor0();
    __syncthreads();
  }

  // ===== epilogue: out = x2 + gamma * (o1 + o2) =====
  float g = gamma[0];
#pragma unroll
  for (int ct = 0; ct < 16; ct++) {
#pragma unroll
    for (int v = 0; v < 8; v++) {
      long long row = (long long)bi * NPIX + qw + v + 8 * h;
      long long o = row * CCH + ct * 16 + m;
      out[o] = x[o * 2 + 1] + g * acc[ct][v];
    }
  }
}

// ---------------- launcher ----------------
extern "C" void kernel_launch(void* const* d_in, const int* in_sizes, int n_in,
                              void* d_out, int out_size, void* d_ws, size_t ws_size,
                              hipStream_t stream) {
  const float* x     = (const float*)d_in[0];
  const float* Wb    = (const float*)d_in[1];
  const float* Wc    = (const float*)d_in[2];
  const float* Wb2   = (const float*)d_in[3];
  const float* Wc2   = (const float*)d_in[4];
  const float* Wd    = (const float*)d_in[5];
  const float* gamma = (const float*)d_in[6];
  unsigned short* ws = (unsigned short*)d_ws;
  float* out = (float*)d_out;

  (void)hipFuncSetAttribute((const void*)proj_kernel,
                            hipFuncAttributeMaxDynamicSharedMemorySize, 196608);
  (void)hipFuncSetAttribute((const void*)attn_kernel,
                            hipFuncAttributeMaxDynamicSharedMemorySize, 98304);

  prep_weights<<<256, 256, 0, stream>>>(Wb, Wc, Wb2, Wc2, Wd, ws);
  proj_kernel <<<256, 256, 196608, stream>>>(x, ws);
  attn_kernel <<<256, 256, 98304,  stream>>>(x, ws, gamma, out);
}
